// LSTMModel_86440511799695
// MI455X (gfx1250) — matrix operand order repaired and checked by reading, not compiled, not run on hardware
//
#include <hip/hip_runtime.h>
#include <hip/hip_bf16.h>

// Problem dims (fixed by reference)
#define LSTM_B   64
#define LSTM_T   128
#define LSTM_E   256
#define LSTM_H   1024
#define LSTM_V   32000
#define LSTM_CAT 1280              // E + H
#define KT_W     (LSTM_CAT / 32)   // 40 K-chunks for recurrent GEMM
#define KT_O     (LSTM_H / 32)     // 32 K-chunks for projection
#define NT_O     (LSTM_V / 16)     // 2000 N-tiles for projection
#define STEP_BLOCKS 16             // persistent kernel blocks (128 waves)

// Element stride between consecutive gate slices in the packed W buffer.
#define GATE_STRIDE ((size_t)64 * KT_W * 512)

typedef __attribute__((ext_vector_type(16))) __bf16 v16bf;
typedef __attribute__((ext_vector_type(8)))  float  v8f;

union V16U { uint4 u[2]; v16bf v; };

__device__ inline v8f vzero8() {
    v8f z = {0.f, 0.f, 0.f, 0.f, 0.f, 0.f, 0.f, 0.f};
    return z;
}

__device__ inline v8f wmma_bf16(v16bf a, v16bf b, v8f c) {
    return __builtin_amdgcn_wmma_f32_16x16x32_bf16(
        false, a, false, b, (short)0, c, false, false);
}

__device__ inline void loadA(V16U& a, const __bf16* p) {
    a.u[0] = *(const uint4*)(p);
    a.u[1] = *(const uint4*)(p + 16);
}
__device__ inline void loadB(V16U& b, const __bf16* p) {
    b.u[0] = ((const uint4*)p)[0];
    b.u[1] = ((const uint4*)p)[1];
}

__device__ inline float fast_sigmoid(float x) {
    return 1.0f / (1.0f + __expf(-x));
}
__device__ inline float fast_tanh(float x) {
    return fmaf(2.0f, fast_sigmoid(2.0f * x), -1.0f);
}

// ---------------------------------------------------------------------------
// Pack a row-major fp32 [K, N] matrix into bf16 WMMA-B tiles.
// Tile (ktile, ntile): tile index = ntile*(K/32) + ktile, 512 bf16 per tile.
// Lane L holds 16 contiguous bf16:
//   lanes 0-15 : N = n0+L,    K = k0 + 0..7, 16..23
//   lanes 16-31: N = n0+L-16, K = k0 + 8..15, 24..31
// ---------------------------------------------------------------------------
__global__ void pack_w_kernel(const float* __restrict__ src,
                              __bf16* __restrict__ dst, int K, int N) {
    size_t total  = (size_t)K * (size_t)N;
    size_t stride = (size_t)gridDim.x * blockDim.x;
    for (size_t i = (size_t)blockIdx.x * blockDim.x + threadIdx.x; i < total;
         i += stride) {
        size_t tile = i >> 9;
        int    r    = (int)(i & 511);
        int    lane = r >> 4;
        int    j    = r & 15;
        int    ktiles = K >> 5;
        size_t ntile  = tile / (size_t)ktiles;
        size_t ktile  = tile - ntile * (size_t)ktiles;
        size_t k = ktile * 32 + (size_t)((j & 8) << 1) + (size_t)((lane & 16) >> 1) + (size_t)(j & 7);
        size_t n = ntile * 16 + (size_t)(lane & 15);
        dst[i] = (__bf16)src[k * (size_t)N + n];
    }
}

__global__ void cvt_x_kernel(const float* __restrict__ src,
                             __bf16* __restrict__ dst, size_t total) {
    size_t stride = (size_t)gridDim.x * blockDim.x;
    for (size_t i = (size_t)blockIdx.x * blockDim.x + threadIdx.x; i < total;
         i += stride)
        dst[i] = (__bf16)src[i];
}

__global__ void init_state_kernel(__bf16* __restrict__ h0,
                                  __bf16* __restrict__ h1,
                                  unsigned* __restrict__ ctr) {
    int i = blockIdx.x * blockDim.x + threadIdx.x;
    if (i < LSTM_B * LSTM_H) {
        h0[i] = (__bf16)0.0f;
        h1[i] = (__bf16)0.0f;
    }
    if (i == 0) *ctr = 0u;
}

// ---------------------------------------------------------------------------
// Persistent LSTM recurrence: ONE launch runs all 128 time steps.
// 128 waves: wave = (mG in 0..1)*64 + (nT in 0..63). Each wave owns a
// 32-row batch group x 16-col H slice for all four gates.
//  - Cell state c stays in 16 VGPRs/lane for the entire sequence.
//  - h ping-pongs between two buffers (read h[t&1], write h[(t+1)&1]) so a
//    single release/acquire grid barrier per step is race-free.
//  - K-loop is double-buffered (loads for kc+1 issued before WMMAs of kc).
// ---------------------------------------------------------------------------
__global__ __launch_bounds__(256, 1) void lstm_persistent_kernel(
    const __bf16* __restrict__ xbf,   // [B, T, E] bf16
    const __bf16* __restrict__ wpk,   // packed W tiles
    const float*  __restrict__ bias,  // [4H]
    __bf16*       __restrict__ hbf0,  // [B, H] ping
    __bf16*       __restrict__ hbf1,  // [B, H] pong
    unsigned*     __restrict__ sync_ctr) {
    const int lane = threadIdx.x & 31;
    const int wave = blockIdx.x * 8 + (threadIdx.x >> 5);
    const int mG   = wave >> 6;   // 0..1 : 32-row batch group
    const int nT   = wave & 63;   // 0..63: 16-col H slice

    const int arow0 = mG * 32 + (lane & 15);
    const int arow1 = arow0 + 16;
    const int koff  = (lane >> 4) << 3;  // 0 or 8

    const __bf16* xrow0 = xbf + (size_t)arow0 * LSTM_T * LSTM_E;
    const __bf16* xrow1 = xbf + (size_t)arow1 * LSTM_T * LSTM_E;
    const __bf16* btile0 = wpk + (size_t)nT * KT_W * 512 + (size_t)lane * 16;

    // Hoisted bias (wave-constant per lane column).
    const int   col = nT * 16 + (lane & 15);
    const float bi  = bias[col];
    const float bj  = bias[LSTM_H + col];
    const float bf_ = bias[2 * LSTM_H + col];
    const float bo  = bias[3 * LSTM_H + col];

    // Cell state in registers for the whole sequence.
    float creg[2][8];
#pragma unroll
    for (int s = 0; s < 2; ++s)
#pragma unroll
        for (int r = 0; r < 8; ++r) creg[s][r] = 0.0f;

    for (int t = 0; t < LSTM_T; ++t) {
        const __bf16* hr = (t & 1) ? hbf1 : hbf0;   // read h(t)
        __bf16*       hw = (t & 1) ? hbf0 : hbf1;   // write h(t+1)
        // Shifted so (hrowX + k0) is valid for k0 in [E, CAT).
        const __bf16* hrow0 = hr + (size_t)arow0 * LSTM_H - LSTM_E;
        const __bf16* hrow1 = hr + (size_t)arow1 * LSTM_H - LSTM_E;

        v8f vz = vzero8();
        v8f acc[4][2] = {{vz, vz}, {vz, vz}, {vz, vz}, {vz, vz}};

        V16U aC[2], aN[2], bC[4], bN[4];
        loadA(aC[0], xrow0 + (size_t)t * LSTM_E + koff);
        loadA(aC[1], xrow1 + (size_t)t * LSTM_E + koff);
#pragma unroll
        for (int g = 0; g < 4; ++g)
            loadB(bC[g], btile0 + (size_t)g * GATE_STRIDE);

#pragma unroll 2
        for (int kc = 0; kc < KT_W; ++kc) {
            const int kn = kc + 1;
            if (kn < KT_W) {
                const int k0 = kn * 32;
                __builtin_prefetch(
                    (const void*)(btile0 + (size_t)(kn + 1) * 512), 0, 1);
                const __bf16* a0 =
                    (k0 < LSTM_E ? xrow0 + (size_t)t * LSTM_E : hrow0) + k0 +
                    koff;
                const __bf16* a1 =
                    (k0 < LSTM_E ? xrow1 + (size_t)t * LSTM_E : hrow1) + k0 +
                    koff;
                loadA(aN[0], a0);
                loadA(aN[1], a1);
#pragma unroll
                for (int g = 0; g < 4; ++g)
                    loadB(bN[g],
                          btile0 + (size_t)g * GATE_STRIDE + (size_t)kn * 512);
            }
#pragma unroll
            for (int g = 0; g < 4; ++g) {
                acc[g][0] = wmma_bf16(aC[0].v, bC[g].v, acc[g][0]);
                acc[g][1] = wmma_bf16(aC[1].v, bC[g].v, acc[g][1]);
            }
            aC[0] = aN[0];
            aC[1] = aN[1];
#pragma unroll
            for (int g = 0; g < 4; ++g) bC[g] = bN[g];
        }

        // Fused gates + in-register cell update; write h(t+1) as bf16.
#pragma unroll
        for (int s = 0; s < 2; ++s) {
            const int rbase = mG * 32 + s * 16 + ((lane >> 4) << 3);
#pragma unroll
            for (int r = 0; r < 8; ++r) {
                const float iv = fast_sigmoid(acc[0][s][r] + bi);
                const float jv = fast_tanh(acc[1][s][r] + bj);
                const float fv = fast_sigmoid(acc[2][s][r] + bf_ + 1.0f);
                const float ov = fast_sigmoid(acc[3][s][r] + bo);
                const float cn = creg[s][r] * fv + iv * jv;
                creg[s][r]     = cn;
                hw[(size_t)(rbase + r) * LSTM_H + col] =
                    (__bf16)(fast_tanh(cn) * ov);
            }
        }

        // Grid-wide release/acquire barrier (16 co-resident blocks).
        __threadfence();
        __syncthreads();
        if (threadIdx.x == 0) {
            const unsigned tgt = (unsigned)STEP_BLOCKS * (unsigned)(t + 1);
            __hip_atomic_fetch_add(sync_ctr, 1u, __ATOMIC_RELEASE,
                                   __HIP_MEMORY_SCOPE_AGENT);
            while (__hip_atomic_load(sync_ctr, __ATOMIC_ACQUIRE,
                                     __HIP_MEMORY_SCOPE_AGENT) < tgt)
                __builtin_amdgcn_s_sleep(1);
        }
        __syncthreads();
    }
}

// ---------------------------------------------------------------------------
// Output projection: out[64, 32000] = h @ w_out + b_out.
// 2000 waves, one per 16-col V slice; each wave computes ALL four 16-row
// batch tiles so w_out is read exactly once (HBM roofline). Double-buffered.
// ---------------------------------------------------------------------------
__global__ __launch_bounds__(256, 1) void lstm_proj_kernel(
    const __bf16* __restrict__ hbf,    // [B, H] (final h)
    const __bf16* __restrict__ wopk,   // packed w_out tiles
    const float*  __restrict__ b_out,  // [V]
    float*        __restrict__ out) {  // [B, V]
    const int lane = threadIdx.x & 31;
    const int nT   = blockIdx.x * 8 + (threadIdx.x >> 5);  // 0..1999

    const int koff = (lane >> 4) << 3;
    const __bf16* hrow[4];
#pragma unroll
    for (int m = 0; m < 4; ++m)
        hrow[m] = hbf + (size_t)(m * 16 + (lane & 15)) * LSTM_H + koff;
    const __bf16* btile = wopk + (size_t)nT * KT_O * 512 + (size_t)lane * 16;

    v8f vz = vzero8();
    v8f acc[4] = {vz, vz, vz, vz};

    V16U aC[4], aN[4], bC, bN;
#pragma unroll
    for (int m = 0; m < 4; ++m) loadA(aC[m], hrow[m]);
    loadB(bC, btile);

#pragma unroll 2
    for (int kc = 0; kc < KT_O; ++kc) {
        const int kn = kc + 1;
        if (kn < KT_O) {
            __builtin_prefetch((const void*)(btile + (size_t)(kn + 1) * 512),
                               0, 1);
#pragma unroll
            for (int m = 0; m < 4; ++m) loadA(aN[m], hrow[m] + kn * 32);
            loadB(bN, btile + (size_t)kn * 512);
        }
#pragma unroll
        for (int m = 0; m < 4; ++m) acc[m] = wmma_bf16(aC[m].v, bC.v, acc[m]);
#pragma unroll
        for (int m = 0; m < 4; ++m) aC[m] = aN[m];
        bC = bN;
    }

    const int   col = nT * 16 + (lane & 15);
    const float bb  = b_out[col];
#pragma unroll
    for (int m = 0; m < 4; ++m) {
        const int rbase = m * 16 + ((lane >> 4) << 3);
#pragma unroll
        for (int r = 0; r < 8; ++r)
            out[(size_t)(rbase + r) * LSTM_V + col] = acc[m][r] + bb;
    }
}

// ---------------------------------------------------------------------------
// Workspace layout (bytes):
//   wpk  @ 0          : 10,485,760   packed W bf16
//   wopk @ 10,485,760 : 65,536,000   packed w_out bf16
//   xbf  @ 76,021,760 :  4,194,304   x bf16
//   h0   @ 80,216,064 :    131,072   hidden ping (bf16)
//   h1   @ 80,347,136 :    131,072   hidden pong (bf16)
//   sync @ 80,478,208 :        256   grid barrier counter
// ---------------------------------------------------------------------------
extern "C" void kernel_launch(void* const* d_in, const int* in_sizes, int n_in,
                              void* d_out, int out_size, void* d_ws,
                              size_t ws_size, hipStream_t stream) {
    (void)in_sizes; (void)n_in; (void)out_size; (void)ws_size;

    const float* x     = (const float*)d_in[0];
    const float* wlstm = (const float*)d_in[1];
    const float* bias  = (const float*)d_in[2];
    const float* w_out = (const float*)d_in[3];
    const float* b_out = (const float*)d_in[4];
    float*       out   = (float*)d_out;

    char*     ws   = (char*)d_ws;
    __bf16*   wpk  = (__bf16*)(ws);
    __bf16*   wopk = (__bf16*)(ws + 10485760);
    __bf16*   xbf  = (__bf16*)(ws + 76021760);
    __bf16*   hbf0 = (__bf16*)(ws + 80216064);
    __bf16*   hbf1 = (__bf16*)(ws + 80347136);
    unsigned* sctr = (unsigned*)(ws + 80478208);

    pack_w_kernel<<<2048, 256, 0, stream>>>(wlstm, wpk, LSTM_CAT, 4 * LSTM_H);
    pack_w_kernel<<<8192, 256, 0, stream>>>(w_out, wopk, LSTM_H, LSTM_V);
    cvt_x_kernel<<<2048, 256, 0, stream>>>(
        x, xbf, (size_t)LSTM_B * LSTM_T * LSTM_E);
    init_state_kernel<<<256, 256, 0, stream>>>(hbf0, hbf1, sctr);

    // Single persistent launch for all 128 time steps.
    lstm_persistent_kernel<<<STEP_BLOCKS, 256, 0, stream>>>(
        xbf, wpk, bias, hbf0, hbf1, sctr);

    // t = 127 writes buffer (127+1)&1 == 0 -> final h is in hbf0.
    lstm_proj_kernel<<<250, 256, 0, stream>>>(hbf0, wopk, b_out, out);
}
